// DGN6_70428873720410
// MI455X (gfx1250) — matrix operand order repaired and checked by reading, not compiled, not run on hardware
//
#include <hip/hip_runtime.h>
#include <cstdint>
#include <cstddef>

typedef __attribute__((ext_vector_type(16))) __bf16 bf16x16;
typedef __attribute__((ext_vector_type(8)))  float  f32x8;
typedef __attribute__((ext_vector_type(4)))  float  f32x4;
typedef __attribute__((ext_vector_type(4)))  unsigned int u32x4;

struct U32x8 { u32x4 lo, hi; };

constexpr int Tdim = 2048;
constexpr int Ddim = 1024;
constexpr int NR   = 16;          // rows per workgroup (one WMMA M-tile)
constexpr int NT   = 4;           // j-tiles per wave pass (N = 64): A-frag reuse x4
constexpr int NW   = 4;           // waves per block (wave32)
constexpr int BLK  = NW * 32;     // 128 threads
constexpr int AST  = Ddim + 8;    // LDS A-strip stride in bf16 elems: 2064B, 16B-aligned, bank-skewed
constexpr int KMAX = 16;

// ---------------- fp32 -> packed bf16 (RNE) ----------------
__global__ __launch_bounds__(256) void cvt_bf16_kernel(
    const float* __restrict__ h, unsigned short* __restrict__ o, int n2)
{
  int i = blockIdx.x * blockDim.x + threadIdx.x;   // one packed u32 (2 elems) per thread
  if (i < n2) {
    unsigned int ua = __float_as_uint(h[2 * i]);
    unsigned int ub = __float_as_uint(h[2 * i + 1]);
    ua += 0x7fffu + ((ua >> 16) & 1u);
    ub += 0x7fffu + ((ub >> 16) & 1u);
    ((unsigned int*)o)[i] = (ua >> 16) | (ub & 0xffff0000u);
  }
}

// ---------------- fused round: scores (WMMA) + causal top-K + gather/mean + update ----------------
__global__ __launch_bounds__(BLK) void dgn_round_kernel(
    const float* __restrict__ hcur,
    const unsigned short* __restrict__ hbf,
    float* __restrict__ hnext,
    const float* __restrict__ gain,
    const float* __restrict__ bias,
    const float* __restrict__ logmix,
    const float* __restrict__ logmom,
    int K)
{
  __shared__ unsigned short abuf[NR * AST];          // A strip: 16 x 1024 bf16 (skewed)
  __shared__ float stile[NW][NR][NR + 1];            // per-wave 16x16 score staging
  __shared__ float wtv[NW][NR][KMAX];                // per-wave per-row top-K values (desc)
  __shared__ int   wti[NW][NR][KMAX];                // per-wave per-row top-K indices
  __shared__ int   topi[NR][KMAX];                   // merged per-row top-K indices
  __shared__ int   topcnt[NR];

  const int tid  = threadIdx.x;
  const int wave = tid >> 5;
  const int lane = tid & 31;
  const int nit  = Tdim / NR;
  const int b    = blockIdx.x / nit;
  const int it   = blockIdx.x % nit;
  const int i0   = it * NR;

  const unsigned short* hb = hbf  + (size_t)b * Tdim * Ddim;
  const float*          hc = hcur + (size_t)b * Tdim * Ddim;
  float*                hn = hnext + (size_t)b * Tdim * Ddim;

  // ---- stage A strip into LDS via async DMA (ASYNCcnt-tracked, per-lane LDS addr keeps skew) ----
  for (int idx = tid; idx < NR * (Ddim / 8); idx += BLK) {
    const int row = idx / (Ddim / 8);
    const int k8  = (idx % (Ddim / 8)) * 8;
    const unsigned short* gp = hb + (size_t)(i0 + row) * Ddim + k8;
    const unsigned lp = (unsigned)(uintptr_t)(&abuf[row * AST + k8]);
    asm volatile("global_load_async_to_lds_b128 %0, %1, off"
                 :: "v"(lp), "v"((unsigned long long)(uintptr_t)gp)
                 : "memory");
  }
  for (int idx = tid; idx < NW * NR * KMAX; idx += BLK) {
    (&wtv[0][0][0])[idx] = -__builtin_inff();
    (&wti[0][0][0])[idx] = 0;
  }
  asm volatile("s_wait_asynccnt 0" ::: "memory");
  __syncthreads();

  const int hlf = lane >> 4;      // 0: lanes 0-15, 1: lanes 16-31 (ISA half-wave split)
  const int l16 = lane & 15;
  const int ntiles = it + 1;      // causal: j-tiles 0..it
  const int ngrp   = (ntiles + NT - 1) / NT;

  for (int g = wave; g < ngrp; g += NW) {
    // B column n = l16 of tile t is row (jt*16 + l16) of h: contiguous K in memory.
    const unsigned short* bcol[NT];
    #pragma unroll
    for (int t = 0; t < NT; ++t) {
      const int jt = g * NT + t;
      const int jrow = (jt < ntiles) ? (jt * NR + l16) : l16;  // clamp dead tiles (masked later)
      bcol[t] = hb + (size_t)jrow * Ddim;
    }
    const unsigned short* arow = &abuf[l16 * AST];   // A row m = l16 in LDS

    f32x8 acc[NT];
    #pragma unroll
    for (int t = 0; t < NT; ++t)
      #pragma unroll
      for (int r = 0; r < 8; ++r) acc[t][r] = 0.0f;

    #pragma unroll 2
    for (int kb = 0; kb < Ddim; kb += 32) {
      __builtin_prefetch(bcol[(kb >> 5) & (NT - 1)] + kb + 256, 0, 3);
      // A frag (shared by all 4 tiles): VGPR0-3 = K kb+{0..7|8..15}, VGPR4-7 = +16, per half
      U32x8 aW;
      aW.lo = *(const u32x4*)(arow + kb + hlf * 8);
      aW.hi = *(const u32x4*)(arow + kb + 16 + hlf * 8);
      const bf16x16 av = __builtin_bit_cast(bf16x16, aW);
      #pragma unroll
      for (int t = 0; t < NT; ++t) {
        // B frag: VGPR0-3 = K kb+{0..7|8..15}, VGPR4-7 = +16, per half (matches A)
        U32x8 bW;
        bW.lo = *(const u32x4*)(bcol[t] + kb + hlf * 8);
        bW.hi = *(const u32x4*)(bcol[t] + kb + 16 + hlf * 8);
        const bf16x16 bv = __builtin_bit_cast(bf16x16, bW);
        acc[t] = __builtin_amdgcn_wmma_f32_16x16x32_bf16(
            false, av, false, bv, (short)0, acc[t], false, false);
      }
    }

    #pragma unroll 1
    for (int t = 0; t < NT; ++t) {
      const int jt = g * NT + t;
      if (jt >= ntiles) break;                        // wave-uniform
      const int j0 = jt * NR;
      // C layout: lanes 0-15 -> (M=r, N=lane), lanes 16-31 -> (M=r+8, N=lane-16)
      #pragma unroll
      for (int r = 0; r < 8; ++r)
        stile[wave][r + hlf * 8][l16] = acc[t][r];
      asm volatile("s_wait_dscnt 0" ::: "memory");    // cross-lane LDS visibility in-wave

      if (lane < 16) {                                // one lane per row: top-K insertion
        const int m = lane;
        const int i = i0 + m;
        float* tv = wtv[wave][m];
        int*   ti = wti[wave][m];
        for (int c = 0; c < NR; ++c) {
          const int j = j0 + c;
          if (j > i) break;                           // causal mask (only diagonal tile trims)
          const float s = stile[wave][m][c];
          if (s > tv[K - 1]) {
            int p = K - 1;
            while (p > 0 && tv[p - 1] < s) {
              tv[p] = tv[p - 1]; ti[p] = ti[p - 1]; --p;
            }
            tv[p] = s; ti[p] = j;
          }
        }
      }
    }
  }
  __syncthreads();

  // ---- merge the 4 per-wave sorted lists -> global top-cnt per row ----
  if (tid < NR) {
    const int m = tid;
    const int i = i0 + m;
    const int cnt = (i + 1 < K) ? (i + 1) : K;       // == reference's valid count
    topcnt[m] = cnt;
    int head[NW] = {0, 0, 0, 0};
    for (int kk = 0; kk < cnt; ++kk) {
      float best = -__builtin_inff(); int bw = 0;
      #pragma unroll
      for (int w = 0; w < NW; ++w) {
        if (head[w] < K) {
          const float v = wtv[w][m][head[w]];
          if (v > best) { best = v; bw = w; }
        }
      }
      topi[m][kk] = wti[bw][m][head[bw]];
      head[bw]++;
    }
  }
  __syncthreads();

  const float mixv = 1.0f / (1.0f + __expf(-logmix[0]));
  const float momv = 1.0f / (1.0f + __expf(-logmom[0]));

  // ---- gather top-K rows (L2-resident), mean, blend, affine, exact GELU, momentum ----
  for (int m = 0; m < NR; ++m) {
    const int cnt = topcnt[m];
    const float inv = 1.0f / (float)cnt;
    const size_t rowoff = (size_t)(i0 + m) * Ddim;
    for (int d = tid * 4; d < Ddim; d += BLK * 4) {
      f32x4 s = {0.0f, 0.0f, 0.0f, 0.0f};
      for (int k = 0; k < cnt; ++k)
        s += *(const f32x4*)(hc + (size_t)topi[m][k] * Ddim + d);
      const f32x4 msg = s * inv;
      const f32x4 hv  = *(const f32x4*)(hc + rowoff + d);
      const f32x4 blended = mixv * hv + (1.0f - mixv) * msg;
      const f32x4 tv = blended * (*(const f32x4*)(gain + d)) + (*(const f32x4*)(bias + d));
      f32x4 outv;
      #pragma unroll
      for (int c = 0; c < 4; ++c) {
        const float t = tv[c];
        const float gl = 0.5f * t * (1.0f + erff(t * 0.70710678f));
        outv[c] = momv * hv[c] + (1.0f - momv) * gl;
      }
      *(f32x4*)(hn + rowoff + d) = outv;
    }
  }
}

// ---------------- output: (h - x) * (softplus(log_scale) + 0.01) ----------------
__global__ __launch_bounds__(256) void finalize_kernel(
    const float* __restrict__ h, const float* __restrict__ x,
    const float* __restrict__ logscale, float* __restrict__ out, int n)
{
  int i = blockIdx.x * blockDim.x + threadIdx.x;
  if (i < n) {
    const float z = logscale[0];
    const float scale = log1pf(__expf(z)) + 0.01f;
    out[i] = (h[i] - x[i]) * scale;
  }
}

extern "C" void kernel_launch(void* const* d_in, const int* in_sizes, int n_in,
                              void* d_out, int out_size, void* d_ws, size_t ws_size,
                              hipStream_t stream) {
  const float* x        = (const float*)d_in[0];
  const float* gain     = (const float*)d_in[1];
  const float* bias     = (const float*)d_in[2];
  const float* logmix   = (const float*)d_in[3];
  const float* logmom   = (const float*)d_in[4];
  const float* logscale = (const float*)d_in[5];

  const int N = in_sizes[0];                // B*T*D = 4,194,304
  const int B = N / (Tdim * Ddim);          // 2

  float* bufA = (float*)d_ws;               // 16 MB
  float* bufB = bufA + N;                   // 16 MB
  unsigned short* hbf = (unsigned short*)(bufB + N);  // 8 MB bf16 mirror

  const int Ks[3] = {4, 8, 16};
  const float* hcur = x;                    // round 0 reads x directly (never written)
  float* hnext = bufA;

  for (int r = 0; r < 3; ++r) {
    const int n2 = N / 2;
    cvt_bf16_kernel<<<dim3((n2 + 255) / 256), dim3(256), 0, stream>>>(hcur, hbf, n2);
    dgn_round_kernel<<<dim3(B * (Tdim / NR)), dim3(BLK), 0, stream>>>(
        hcur, hbf, hnext,
        gain + r * Ddim, bias + r * Ddim, logmix + r, logmom, Ks[r]);
    hcur  = hnext;
    hnext = (hcur == bufA) ? bufB : bufA;
  }
  finalize_kernel<<<dim3((N + 255) / 256), dim3(256), 0, stream>>>(
      hcur, x, logscale, (float*)d_out, N);
}
